// ELISSABETH_20143396618287
// MI455X (gfx1250) — matrix operand order repaired and checked by reading, not compiled, not run on hardware
//
#include <hip/hip_runtime.h>
#include <hip/hip_bf16.h>
#include <math.h>

// ---------------------------------------------------------------------------
// ELISSABETH (LISS) model on MI455X / gfx1250, wave32 + WMMA + async-to-LDS.
//   B=2, T=8192, D=128, H=64, L=3, NL=4, VOUT=8192
// Roofline: 512MB transposed logits write (~22us @ 23.3TB/s) dominates; the
// 3-stage exponential scan is parallelized per (b,h) channel with a block
// prefix-scan in LDS so it no longer serializes 8192 steps on 128 threads.
// ---------------------------------------------------------------------------

#define TT     8192
#define BBATCH 2
#define DD     128
#define HH     64
#define LLISS  3
#define NLAYER 4
#define VOUTN  8192
#define MROWS  (BBATCH * TT)          // 16384

typedef __attribute__((ext_vector_type(16))) _Float16 v16h;
typedef __attribute__((ext_vector_type(8)))  float    v8f;
typedef __attribute__((ext_vector_type(4)))  float    f32x4;
typedef __attribute__((ext_vector_type(4)))  int      v4i;

typedef __attribute__((address_space(1))) v4i* gv4i_p;   // global int4*
typedef __attribute__((address_space(3))) v4i* lv4i_p;   // LDS int4*

// ---------------------------------------------------------------------------
// gfx1250 async global->LDS copy (16B per lane), graceful fallback.
// Builtin signature (from clang diagnostic): (AS1 v4i*, AS3 v4i*, Imm, Imm).
// ---------------------------------------------------------------------------
__device__ __forceinline__ void cp16_async(float* lds, const float* g)
{
#if __has_builtin(__builtin_amdgcn_global_load_async_to_lds_b128)
    __builtin_amdgcn_global_load_async_to_lds_b128(
        (gv4i_p)(g), (lv4i_p)(lds), 0, 0);
#else
    *(f32x4*)lds = *(const f32x4*)g;
#endif
}

__device__ __forceinline__ void cp_async_join()
{
#if __has_builtin(__builtin_amdgcn_global_load_async_to_lds_b128)
#  if __has_builtin(__builtin_amdgcn_s_wait_asynccnt)
    __builtin_amdgcn_s_wait_asynccnt(0);
#  else
    asm volatile("s_wait_asynccnt 0x0" ::: "memory");
#  endif
#endif
    __syncthreads();
}

// ---------------------------------------------------------------------------
// WMMA K-loop from global memory (used by the small/thin GEMMs).
// Fragments use contiguous k = 16*half + e in both A and B (16-bit A 16x32, B 32x16).
// ---------------------------------------------------------------------------
__device__ __forceinline__ v8f wmma_kloop(const float* __restrict__ A, int lda,
                                          const float* __restrict__ B, int ldb,
                                          int m0, int n0, int K, v8f acc)
{
    const int lane = threadIdx.x & 31;
    const int lq   = lane & 15;
    const int half = lane >> 4;

    const float* __restrict__ arow = A + (size_t)(m0 + lq) * (size_t)lda;

    for (int kb = 0; kb < K; kb += 32) {
        v16h a, b;
        #pragma unroll
        for (int e = 0; e < 16; ++e) {
            a[e] = (_Float16)arow[kb + half * 16 + e];
        }
        const float* __restrict__ bcol =
            B + (size_t)(kb + half * 16) * (size_t)ldb + (size_t)(n0 + lq);
        __builtin_prefetch(bcol + (size_t)32 * (size_t)ldb, 0, 1); // global_prefetch_b8
        #pragma unroll
        for (int e = 0; e < 16; ++e) {
            b[e] = (_Float16)bcol[(size_t)e * (size_t)ldb];
        }
        acc = __builtin_amdgcn_wmma_f32_16x16x32_f16(
            false, a, false, b, (short)0, acc, false, false);
    }
    return acc;
}

// ---------------------------------------------------------------------------
// Generic tiled GEMM, 4 waves/block, one 16x16 tile per wave along N.
// TRANSC=false: C[m,n] row-major (ldc).  TRANSC=true: C[n,m] (ldc = M stride),
// with two b128 stores per lane (8 consecutive m per lane).
// ---------------------------------------------------------------------------
template <bool TRANSC>
__global__ void gemm16_kernel(const float* __restrict__ A, int lda,
                              const float* __restrict__ B, int ldb,
                              float* __restrict__ C, int ldc, int K)
{
    const int wave = threadIdx.x >> 5;
    const int m0   = blockIdx.y * 16;
    const int n0   = (blockIdx.x * 4 + wave) * 16;

    v8f acc = {};
    acc = wmma_kloop(A, lda, B, ldb, m0, n0, K, acc);

    const int lane = threadIdx.x & 31;
    const int lq   = lane & 15;
    const int half = lane >> 4;
    const int n    = n0 + lq;

    if (TRANSC) {
        float* base = C + (size_t)n * (size_t)ldc + (size_t)(m0 + half * 8);
        f32x4 lo = { acc[0], acc[1], acc[2], acc[3] };
        f32x4 hi = { acc[4], acc[5], acc[6], acc[7] };
        *(f32x4*)(base)     = lo;
        *(f32x4*)(base + 4) = hi;
    } else {
        #pragma unroll
        for (int r = 0; r < 8; ++r) {
            const int m = m0 + half * 8 + r;
            C[(size_t)m * (size_t)ldc + (size_t)n] = acc[r];
        }
    }
}

// ---------------------------------------------------------------------------
// Unembed: logits = X(16384x128) @ W_un(128x8192) + b_un, transposed store
// out[b, n, t].  Block = 256 thr (8 waves) -> 32(M) x 64(N) tile.
// A (32x128, padded stride 132 to avoid bank conflicts) and B (128x64) are
// staged into LDS with async b128 copies, fragments then read via ds_load.
// ---------------------------------------------------------------------------
#define SA_STRIDE 132

__global__ void unembed_kernel(const float* __restrict__ X,
                               const float* __restrict__ Wun,
                               const float* __restrict__ bun,
                               float* __restrict__ out)
{
    __shared__ float sA[32 * SA_STRIDE];  // 16.5 KB
    __shared__ float sB[128 * 64];        // 32   KB

    const int tid    = threadIdx.x;       // 0..255
    const int m0blk  = blockIdx.y * 32;
    const int n0blk  = blockIdx.x * 64;

    // Stage A: 32 rows x 128 f32 = 1024 x b128 transfers (4 per thread)
    for (int ti = tid; ti < 1024; ti += 256) {
        const int row = ti >> 5;          // 32 float4 per row
        const int c4  = (ti & 31) * 4;
        cp16_async(&sA[row * SA_STRIDE + c4],
                   X + (size_t)(m0blk + row) * DD + c4);
    }
    // Stage B: 128 rows x 64 f32 = 2048 x b128 transfers (8 per thread)
    for (int ti = tid; ti < 2048; ti += 256) {
        const int row = ti >> 4;          // 16 float4 per row
        const int c4  = (ti & 15) * 4;
        cp16_async(&sB[row * 64 + c4],
                   Wun + (size_t)row * VOUTN + (size_t)n0blk + c4);
    }
    cp_async_join();

    const int wave = tid >> 5;
    const int mi   = wave >> 2;           // 0..1
    const int ni   = wave & 3;            // 0..3
    const int lane = tid & 31;
    const int lq   = lane & 15;
    const int half = lane >> 4;
    const int lm   = mi * 16 + lq;        // local A row
    const int ln   = ni * 16 + lq;        // local B col

    v8f acc = {};
    #pragma unroll
    for (int kb = 0; kb < DD; kb += 32) {
        v16h a, b;
        #pragma unroll
        for (int e = 0; e < 16; ++e) {
            a[e] = (_Float16)sA[lm * SA_STRIDE + kb + half * 16 + e];
        }
        #pragma unroll
        for (int e = 0; e < 16; ++e) {
            b[e] = (_Float16)sB[(kb + half * 16 + e) * 64 + ln];
        }
        acc = __builtin_amdgcn_wmma_f32_16x16x32_f16(
            false, a, false, b, (short)0, acc, false, false);
    }

    const int n    = n0blk + ln;
    const int m0   = m0blk + mi * 16;
    const float bias = bun[n];
    #pragma unroll
    for (int r = 0; r < 8; ++r) {
        const int m = m0 + half * 8 + r;
        const int b = m >> 13;            // T = 8192
        const int t = m & (TT - 1);
        out[((size_t)b * VOUTN + (size_t)n) * TT + (size_t)t] = acc[r] + bias;
    }
}

// ---------------------------------------------------------------------------
// Embedding gather: x[bt, d] = emb[tokens[bt], d]
// ---------------------------------------------------------------------------
__global__ void embed_kernel(const int* __restrict__ tokens,
                             const float* __restrict__ emb,
                             float* __restrict__ x)
{
    const int bt  = blockIdx.x;
    const int d   = threadIdx.x;
    const int tok = tokens[bt];
    x[(size_t)bt * DD + d] = emb[(size_t)tok * DD + d];
}

// ---------------------------------------------------------------------------
// Parallel fused 3-stage LISS scan.  One 256-thread block per (b,h) channel.
// q/k/v come in TRANSPOSED (L*H, B*T) layout -> each thread streams E=32
// contiguous t values.  Each stage: thread-local inclusive prefix, block scan
// of 256 partials (Hillis-Steele in LDS), res_l kept in a 32KB LDS buffer so
// stage l+1 can read res_l[t-1] in-place (reads before writes, barriered).
// ---------------------------------------------------------------------------
#define SCAN_NT 256
#define SCAN_E  (TT / SCAN_NT)          // 32

__global__ void liss_scan_kernel(const float* __restrict__ qT,
                                 const float* __restrict__ kT,
                                 const float* __restrict__ vT,
                                 float* __restrict__ res)
{
    __shared__ float resbuf[TT];        // 32 KB: res_l for the channel
    __shared__ float sums[SCAN_NT];

    const int bh = blockIdx.x;          // 0..B*H-1
    const int b  = bh >> 6;             // / H
    const int h  = bh & (HH - 1);
    const int j  = threadIdx.x;
    const int t0 = j * SCAN_E;

    const float* __restrict__ qs[LLISS];
    const float* __restrict__ ks[LLISS];
    const float* __restrict__ vs[LLISS];
    #pragma unroll
    for (int l = 0; l < LLISS; ++l) {
        const size_t off = (size_t)(l * HH + h) * MROWS + (size_t)b * TT;
        qs[l] = qT + off;
        ks[l] = kT + off;
        vs[l] = vT + off;
    }

    float a[SCAN_E];

    #pragma unroll
    for (int l = 0; l < LLISS; ++l) {
        // phase 1: elementwise term + thread-local inclusive prefix
        float run = 0.f;
        if (l == 0) {
            for (int i = 0; i < SCAN_E; ++i) {
                const int t = t0 + i;
                run += expf(-ks[0][t]) * vs[0][t];
                a[i] = run;
            }
        } else {
            for (int i = 0; i < SCAN_E; ++i) {
                const int t  = t0 + i;
                const float rp = (t == 0) ? 0.f : resbuf[t - 1];   // res_{l-1}[t-1]
                run += expf(qs[l - 1][t] - ks[l][t]) * vs[l][t] * rp;
                a[i] = run;
            }
        }
        __syncthreads();                 // all resbuf reads complete

        // phase 2: block-wide inclusive scan of per-thread sums
        sums[j] = run;
        __syncthreads();
        float incl = run;
        for (int off = 1; off < SCAN_NT; off <<= 1) {
            const float add = (j >= off) ? sums[j - off] : 0.f;
            __syncthreads();
            incl += add;
            sums[j] = incl;
            __syncthreads();
        }
        const float offx = incl - run;   // exclusive offset for this thread

        // phase 3: publish res_l
        for (int i = 0; i < SCAN_E; ++i)
            resbuf[t0 + i] = offx + a[i];
        __syncthreads();
    }

    // final gating: res[t] = exp(q_{L-1}[t]) * res_{L-1}[t], row-major (m, h)
    for (int i = 0; i < SCAN_E; ++i) {
        const int t = t0 + i;
        res[((size_t)b * TT + t) * HH + h] = expf(qs[LLISS - 1][t]) * resbuf[t];
    }
}

// ---------------------------------------------------------------------------
// Orchestration
// ---------------------------------------------------------------------------
extern "C" void kernel_launch(void* const* d_in, const int* in_sizes, int n_in,
                              void* d_out, int out_size, void* d_ws, size_t ws_size,
                              hipStream_t stream)
{
    (void)in_sizes; (void)n_in; (void)out_size; (void)ws_size;

    const int*   tokens = (const int*)  d_in[0];
    const float* emb    = (const float*)d_in[1];
    const float* WQ     = (const float*)d_in[2];   // (NL, L, D, H)
    const float* WK     = (const float*)d_in[3];
    const float* WV     = (const float*)d_in[4];
    const float* WO     = (const float*)d_in[5];   // (NL, H, D)
    const float* Wun    = (const float*)d_in[6];   // (D, VOUT)
    const float* bun    = (const float*)d_in[7];   // (VOUT,)
    float*       out    = (float*)d_out;           // (B, VOUT, T)

    const size_t M = (size_t)MROWS;

    // Workspace: 2*8MB x + 3*12MB qkv(T) + 4MB res = 56MB
    float* x0  = (float*)d_ws;
    float* x1  = x0 + M * DD;
    float* qT  = x1 + M * DD;                 // (L*H, M)
    float* kT  = qT + (size_t)LLISS * HH * M;
    float* vT  = kT + (size_t)LLISS * HH * M;
    float* res = vT + (size_t)LLISS * HH * M; // (M, H)

    embed_kernel<<<dim3((unsigned)M), dim3(DD), 0, stream>>>(tokens, emb, x0);

    float* xin  = x0;
    float* xout = x1;

    const dim3 blk4(128);                                  // 4 waves
    const dim3 grid_qkv(HH / 64, (unsigned)(M / 16));      // N=64
    const dim3 grid_proj(DD / 64, (unsigned)(M / 16));     // N=128
    const dim3 grid_un(VOUTN / 64, (unsigned)(M / 32));    // 32x64 block tiles

    for (int i = 0; i < NLAYER; ++i) {
        // q,k,v projections, stored transposed (L*H, M) for the scan
        for (int l = 0; l < LLISS; ++l) {
            const size_t woff = ((size_t)i * LLISS + l) * DD * HH;
            const size_t coff = (size_t)l * HH * M;
            gemm16_kernel<true><<<grid_qkv, blk4, 0, stream>>>(
                xin, DD, WQ + woff, HH, qT + coff, (int)M, DD);
            gemm16_kernel<true><<<grid_qkv, blk4, 0, stream>>>(
                xin, DD, WK + woff, HH, kT + coff, (int)M, DD);
            gemm16_kernel<true><<<grid_qkv, blk4, 0, stream>>>(
                xin, DD, WV + woff, HH, vT + coff, (int)M, DD);
        }
        // block-parallel fused scan: one workgroup per (b,h) channel
        liss_scan_kernel<<<dim3(BBATCH * HH), dim3(SCAN_NT), 0, stream>>>(
            qT, kT, vT, res);
        // output projection: (16384 x 64) @ (64 x 128), row-major
        gemm16_kernel<false><<<grid_proj, blk4, 0, stream>>>(
            res, HH, WO + (size_t)i * HH * DD, DD, xout, DD, HH);
        float* tmp = xin; xin = xout; xout = tmp;
    }

    unembed_kernel<<<grid_un, dim3(256), 0, stream>>>(xin, Wun, bun, out);
}
